// MultiHeadSelfAttention_87806311399977
// MI455X (gfx1250) — matrix operand order repaired and checked by reading, not compiled, not run on hardware
//
#include <hip/hip_runtime.h>
#include <hip/hip_bf16.h>

typedef _Float16 h16;
typedef __attribute__((ext_vector_type(16))) _Float16 v16h;
typedef __attribute__((ext_vector_type(8)))  _Float16 v8h;
typedef __attribute__((ext_vector_type(8)))  float    v8f;
typedef __attribute__((ext_vector_type(4)))  unsigned int u32x4;
typedef __attribute__((ext_vector_type(8)))  unsigned int u32x8;

#define DIM 1024
#define SEQ 2048
#define NH  16
#define HD  64

__device__ __forceinline__ v16h load_frag(const h16* p0, const h16* p1) {
    union { v16h v; v8h h[2]; } u;
    u.h[0] = *(const v8h*)p0;
    u.h[1] = *(const v8h*)p1;
    return u.v;
}

__device__ __forceinline__ v8f wmma_f16(v16h a, v16h b, v8f c) {
    return __builtin_amdgcn_wmma_f32_16x16x32_f16(false, a, false, b, (short)0, c,
                                                  false, false);
}

// Low 32 bits of a flat shared-aperture address are the LDS byte offset.
__device__ __forceinline__ unsigned lds_off(const void* p) {
    return (unsigned)(uintptr_t)p;
}

// TDM 2D tile load: rows x cols elements of 2 bytes, global row stride
// `stride` elements, packed contiguously into LDS at `ldsoff`.
// D# built per CDNA5 ISA; VADDR2/3 = NULL => 2D form.
__device__ __forceinline__ void tdm_load_2d(unsigned ldsoff, const h16* gptr,
                                            unsigned rows, unsigned cols,
                                            unsigned stride) {
    unsigned long long ga = (unsigned long long)(uintptr_t)gptr;
    u32x4 g0;
    g0[0] = 1u;                                    // count=1, user mode, no gather
    g0[1] = ldsoff;                                // lds_addr
    g0[2] = (unsigned)(ga & 0xFFFFFFFFu);          // global_addr[31:0]
    g0[3] = (unsigned)((ga >> 32) & 0x01FFFFFFu)   // global_addr[56:32]
          | (2u << 30);                            // type=2 ("image")
    u32x8 g1;
    g1[0] = 1u << 16;                              // data_size=1 (2 bytes)
    g1[1] = (cols & 0xFFFFu) << 16;                // tensor_dim0[15:0]
    g1[2] = (rows & 0xFFFFu) << 16;                // tensor_dim0 hi=0, tensor_dim1 lo
    g1[3] = (cols & 0xFFFFu) << 16;                // tensor_dim1 hi=0, tile_dim0
    g1[4] = rows & 0xFFFFu;                        // tile_dim1; tile_dim2=0
    g1[5] = stride;                                // tensor_dim0_stride[31:0]
    g1[6] = 0u;                                    // stride hi; dim1_stride lo
    g1[7] = 0u;
    asm volatile("tensor_load_to_lds %0, %1" :: "s"(g0), "s"(g1) : "memory");
}

// ---------------------------------------------------------------- fp32 -> f16
__global__ __launch_bounds__(256) void cvt_f32_f16(const float* __restrict__ in,
                                                   h16* __restrict__ out, int n) {
    int i = (blockIdx.x * 256 + threadIdx.x) * 4;
    if (i < n) {
        float4 v = *(const float4*)(in + i);
        out[i + 0] = (h16)v.x;
        out[i + 1] = (h16)v.y;
        out[i + 2] = (h16)v.z;
        out[i + 3] = (h16)v.w;
    }
}

// ------------------------------------------------- GEMM: C = A * B^T + bias
// A: [M,K] f16 row-major, B: [N,K] f16 row-major (weights, so A@B^T).
// mode 0: store f16 row-major [M,N]
// mode 1: store f16 "V-transposed": out[(b*DIM + n)*SEQ + s], row = b*SEQ + s
// mode 2: store f32 row-major [M,N]
__global__ __launch_bounds__(256) void gemm_wmma(const h16* __restrict__ A,
                                                 const h16* __restrict__ B,
                                                 const float* __restrict__ bias,
                                                 void* __restrict__ out,
                                                 int M, int N, int K, int mode) {
    const int nb = N >> 7;
    const int bx = blockIdx.x % nb, by = blockIdx.x / nb;
    const int m0 = by << 7, n0 = bx << 7;
    const int tid = threadIdx.x;
    const int wave = tid >> 5, lane = tid & 31;
    const int wm = wave & 3, wn = wave >> 2;   // 4 x 2 wave grid
    const int l16 = lane & 15, lh = lane >> 4;

    v8f acc[2][4] = {};

    const h16* Ap[2];
    const h16* Bp[4];
#pragma unroll
    for (int mi = 0; mi < 2; ++mi)
        Ap[mi] = A + (size_t)(m0 + wm * 32 + mi * 16 + l16) * K + lh * 8;
#pragma unroll
    for (int nj = 0; nj < 4; ++nj)
        Bp[nj] = B + (size_t)(n0 + wn * 64 + nj * 16 + l16) * K + lh * 8;

    for (int kb = 0; kb < K; kb += 32) {
        v16h a[2], b[4];
#pragma unroll
        for (int mi = 0; mi < 2; ++mi)
            a[mi] = load_frag(Ap[mi] + kb, Ap[mi] + kb + 16);
#pragma unroll
        for (int nj = 0; nj < 4; ++nj)
            b[nj] = load_frag(Bp[nj] + kb, Bp[nj] + kb + 16);
#pragma unroll
        for (int mi = 0; mi < 2; ++mi)
#pragma unroll
            for (int nj = 0; nj < 4; ++nj)
                acc[mi][nj] = wmma_f16(a[mi], b[nj], acc[mi][nj]);
    }

#pragma unroll
    for (int mi = 0; mi < 2; ++mi) {
#pragma unroll
        for (int nj = 0; nj < 4; ++nj) {
            const int col = n0 + wn * 64 + nj * 16 + l16;
            const float bv = bias[col];
#pragma unroll
            for (int r = 0; r < 8; ++r) {
                const int row = m0 + wm * 32 + mi * 16 + r + 8 * lh;
                const float val = acc[mi][nj][r] + bv;
                if (mode == 0) {
                    ((h16*)out)[(size_t)row * N + col] = (h16)val;
                } else if (mode == 1) {
                    const int bb = row >> 11, s = row & (SEQ - 1);
                    ((h16*)out)[((size_t)(bb * DIM + col) << 11) + s] = (h16)val;
                } else {
                    ((float*)out)[(size_t)row * N + col] = val;
                }
            }
        }
    }
}

// ------------------------------------------------------------ flash attention
// Q,K: f16 [B*SEQ, DIM] (token-major, head cols h*64..h*64+63)
// Vt : f16 [(b*NH+h)*64 + d][SEQ]
// O  : f16 [B*SEQ, DIM]
// K/V tiles are staged into LDS by the Tensor Data Mover (one wave issues,
// double-buffered, TENSORcnt-fenced); all 8 waves consume via ds_load_b128.
__global__ __launch_bounds__(256) void flash_attn(const h16* __restrict__ Q,
                                                  const h16* __restrict__ Km,
                                                  const h16* __restrict__ Vt,
                                                  h16* __restrict__ O) {
    const int tid = threadIdx.x;
    const int wave = tid >> 5, lane = tid & 31;
    const int l16 = lane & 15, lh = lane >> 4;
    const int qblk = blockIdx.x & 7;
    const int bh   = blockIdx.x >> 3;
    const int b = bh >> 4, hh = bh & 15;
    const int q0 = qblk * 256 + wave * 32;        // query row in sequence
    const size_t rowbase = (size_t)b * SEQ;
    const int hc = hh * HD;

    __shared__ __align__(16) h16 kt[2][32][64];   // K tile: [key][d]
    __shared__ __align__(16) h16 vt[2][64][32];   // V tile: [d][key]
    __shared__ __align__(16) h16 pt[8][32][40];   // per-wave P tile (+pad)

    const h16* KbaseG = Km + rowbase * DIM + hc;          // K[token][d]
    const h16* VbaseG = Vt + (size_t)bh * HD * SEQ;       // Vt[d][s]

    // Q fragments [mi][dk]: A-layout, contraction d in two chunks of 32
    v16h qf[2][2];
#pragma unroll
    for (int mi = 0; mi < 2; ++mi)
#pragma unroll
        for (int dk = 0; dk < 2; ++dk) {
            const h16* p = Q + (rowbase + q0 + mi * 16 + l16) * DIM + hc + dk * 32 + lh * 8;
            qf[mi][dk] = load_frag(p, p + 16);
        }

    v8f oacc[2][4] = {};
    float m_[2][8], l_[2][8];
#pragma unroll
    for (int mi = 0; mi < 2; ++mi)
#pragma unroll
        for (int r = 0; r < 8; ++r) { m_[mi][r] = -1e30f; l_[mi][r] = 0.f; }

    const float inv_scale = 0.125f;               // 1/sqrt(64)

    // preload tile 0 via TDM
    if (wave == 0) {
        tdm_load_2d(lds_off(&kt[0][0][0]), KbaseG, 32, HD, DIM);
        tdm_load_2d(lds_off(&vt[0][0][0]), VbaseG, HD, 32, SEQ);
        __builtin_amdgcn_s_wait_tensorcnt(0);
    }
    __syncthreads();

    for (int kb = 0; kb < SEQ; kb += 32) {
        const int cur = (kb >> 5) & 1, nxt = cur ^ 1;
        if (wave == 0 && kb + 32 < SEQ) {
            tdm_load_2d(lds_off(&kt[nxt][0][0]), KbaseG + (size_t)(kb + 32) * DIM,
                        32, HD, DIM);
            tdm_load_2d(lds_off(&vt[nxt][0][0]), VbaseG + (kb + 32),
                        HD, 32, SEQ);
        }

        // ---- scores S = Q K^T (32q x 32k), K from LDS
        v8f s[2][2] = {};
#pragma unroll
        for (int dk = 0; dk < 2; ++dk) {
            v16h bk[2];
#pragma unroll
            for (int nj = 0; nj < 2; ++nj) {
                const h16* p = &kt[cur][nj * 16 + l16][dk * 32 + lh * 8];
                bk[nj] = load_frag(p, p + 16);
            }
#pragma unroll
            for (int mi = 0; mi < 2; ++mi)
#pragma unroll
                for (int nj = 0; nj < 2; ++nj)
                    s[mi][nj] = wmma_f16(qf[mi][dk], bk[nj], s[mi][nj]);
        }

        // ---- online softmax, write P (f16) to per-wave LDS tile
#pragma unroll
        for (int mi = 0; mi < 2; ++mi) {
#pragma unroll
            for (int r = 0; r < 8; ++r) {
                float s0 = s[mi][0][r] * inv_scale;
                float s1 = s[mi][1][r] * inv_scale;
                float mx = fmaxf(s0, s1);
#pragma unroll
                for (int off = 8; off >= 1; off >>= 1)
                    mx = fmaxf(mx, __shfl_xor(mx, off, 32));
                const float mnew = fmaxf(m_[mi][r], mx);
                const float alpha = __expf(m_[mi][r] - mnew);
                m_[mi][r] = mnew;
                const float p0 = __expf(s0 - mnew);
                const float p1 = __expf(s1 - mnew);
                s[mi][0][r] = p0;
                s[mi][1][r] = p1;
                float rs = p0 + p1;
#pragma unroll
                for (int off = 8; off >= 1; off >>= 1)
                    rs += __shfl_xor(rs, off, 32);
                l_[mi][r] = l_[mi][r] * alpha + rs;
#pragma unroll
                for (int nj = 0; nj < 4; ++nj)
                    oacc[mi][nj][r] *= alpha;
            }
#pragma unroll
            for (int nj = 0; nj < 2; ++nj)
#pragma unroll
                for (int r = 0; r < 8; ++r)
                    pt[wave][mi * 16 + r + 8 * lh][nj * 16 + l16] = (h16)s[mi][nj][r];
        }

        // ---- O += P V  (contraction = 32 keys), P and V from LDS
        v16h pf[2];
#pragma unroll
        for (int mi = 0; mi < 2; ++mi) {
            const h16* p = &pt[wave][mi * 16 + l16][lh * 8];
            pf[mi] = load_frag(p, p + 16);
        }
#pragma unroll
        for (int nj = 0; nj < 4; ++nj) {
            const h16* p = &vt[cur][nj * 16 + l16][lh * 8];
            const v16h bv = load_frag(p, p + 16);
#pragma unroll
            for (int mi = 0; mi < 2; ++mi)
                oacc[mi][nj] = wmma_f16(pf[mi], bv, oacc[mi][nj]);
        }

        // all waves done reading buf[cur]; ensure buf[nxt] DMA complete
        __syncthreads();
        if (wave == 0)
            __builtin_amdgcn_s_wait_tensorcnt(0);
        __syncthreads();
    }

    // ---- epilogue: O /= l, store f16 token-major
#pragma unroll
    for (int mi = 0; mi < 2; ++mi)
#pragma unroll
        for (int r = 0; r < 8; ++r) {
            const float inv = 1.0f / l_[mi][r];
            const int row = q0 + mi * 16 + r + 8 * lh;
#pragma unroll
            for (int nj = 0; nj < 4; ++nj) {
                const int col = hc + nj * 16 + l16;
                O[(rowbase + row) * DIM + col] = (h16)(oacc[mi][nj][r] * inv);
            }
        }
}

// ---------------------------------------------------------------------- host
extern "C" void kernel_launch(void* const* d_in, const int* in_sizes, int n_in,
                              void* d_out, int out_size, void* d_ws, size_t ws_size,
                              hipStream_t stream) {
    const float* x  = (const float*)d_in[0];
    const float* Wq = (const float*)d_in[1];
    const float* bq = (const float*)d_in[2];
    const float* Wk = (const float*)d_in[3];
    const float* bk = (const float*)d_in[4];
    const float* Wv = (const float*)d_in[5];
    const float* bv = (const float*)d_in[6];
    const float* Wo = (const float*)d_in[7];
    const float* bo = (const float*)d_in[8];

    const int M = 4 * SEQ;         // 8192 tokens
    const int N = DIM, K = DIM;

    char* ws = (char*)d_ws;
    const size_t XH  = (size_t)M * DIM * 2;     // 16 MB
    const size_t WH  = (size_t)DIM * DIM * 2;   //  2 MB
    h16* xh  = (h16*)(ws);
    h16* Wqh = (h16*)(ws + XH);
    h16* Wkh = (h16*)(ws + XH + WH);
    h16* Wvh = (h16*)(ws + XH + 2 * WH);
    h16* Woh = (h16*)(ws + XH + 3 * WH);
    h16* Qh  = (h16*)(ws + XH + 4 * WH);
    h16* Kh  = (h16*)(ws + 2 * XH + 4 * WH);
    h16* Vth = (h16*)(ws + 3 * XH + 4 * WH);
    h16* Oh  = (h16*)(ws + 4 * XH + 4 * WH);

    // fp32 -> f16
    cvt_f32_f16<<<(M * DIM) / 1024, 256, 0, stream>>>(x, xh, M * DIM);
    cvt_f32_f16<<<(DIM * DIM) / 1024, 256, 0, stream>>>(Wq, Wqh, DIM * DIM);
    cvt_f32_f16<<<(DIM * DIM) / 1024, 256, 0, stream>>>(Wk, Wkh, DIM * DIM);
    cvt_f32_f16<<<(DIM * DIM) / 1024, 256, 0, stream>>>(Wv, Wvh, DIM * DIM);
    cvt_f32_f16<<<(DIM * DIM) / 1024, 256, 0, stream>>>(Wo, Woh, DIM * DIM);

    const int gemm_grid = (M / 128) * (N / 128);   // 512
    // Q = x Wq^T + bq ; K = x Wk^T + bk ; Vt = transpose(x Wv^T + bv)
    gemm_wmma<<<gemm_grid, 256, 0, stream>>>(xh, Wqh, bq, Qh,  M, N, K, 0);
    gemm_wmma<<<gemm_grid, 256, 0, stream>>>(xh, Wkh, bk, Kh,  M, N, K, 0);
    gemm_wmma<<<gemm_grid, 256, 0, stream>>>(xh, Wvh, bv, Vth, M, N, K, 1);

    // flash attention: 64 (b,h) pairs x 8 query blocks of 256 rows
    flash_attn<<<64 * 8, 256, 0, stream>>>(Qh, Kh, Vth, Oh);

    // out = O Wo^T + bo  (fp32)
    gemm_wmma<<<gemm_grid, 256, 0, stream>>>(Oh, Woh, bo, d_out, M, N, K, 2);
}
